// StabilityAugmentedMemory_8701603742407
// MI455X (gfx1250) — matrix operand order repaired and checked by reading, not compiled, not run on hardware
//
#include <hip/hip_runtime.h>

// ---------------------------------------------------------------------------
// StabilityAugmentedMemory for MI455X (gfx1250, wave32, WMMA bf16)
// ---------------------------------------------------------------------------

typedef __attribute__((ext_vector_type(16))) __bf16       v16bf;
typedef __attribute__((ext_vector_type(8)))  float        v8f;
typedef __attribute__((ext_vector_type(4)))  float        v4f;
typedef __attribute__((ext_vector_type(2)))  float        v2f;
typedef __attribute__((ext_vector_type(4)))  unsigned int v4u;

union BF16Frag { v4u u[2]; v16bf v; };   // 32 bytes = 16 bf16 = 8 VGPRs

#define D_       128
#define T_       64
#define KP_      5
#define KFOLD    256          // folded K dimension (raw 128 | edge 64 | tenc 64)
#define ASTRIDE  264          // bf16 elements per A/W row (256 + 8 pad, bank skew)
#define QSTRIDE  132          // f32 elements per Q row (128 + 4 pad)
#define WAVE_BYTES 8448       // max(16*264*2, 16*132*4) = 8448
#define WAVES_PER_BLOCK 4
#define WF_LDS_BYTES (D_ * ASTRIDE * 2)   // 128*264*2 = 67584
#define LN_EPS   1e-5f
#define EPS_     1e-12f

__device__ __forceinline__ unsigned short f2bf(float f) {
  unsigned int u = __float_as_uint(f);
  u += 0x7FFFu + ((u >> 16) & 1u);     // round-to-nearest-even
  return (unsigned short)(u >> 16);
}

__device__ __forceinline__ float wsum32(float x) {
  x += __shfl_xor(x, 16, 32);
  x += __shfl_xor(x, 8, 32);
  x += __shfl_xor(x, 4, 32);
  x += __shfl_xor(x, 2, 32);
  x += __shfl_xor(x, 1, 32);
  return x;
}

// --------------------------- kernel 1: memory copy -------------------------
__global__ void sam_copy_kernel(const v4f* __restrict__ in, v4f* __restrict__ out, int n4) {
  int i = blockIdx.x * blockDim.x + threadIdx.x;
  int stride = gridDim.x * blockDim.x;
  for (; i < n4; i += stride) out[i] = in[i];
}

// ------------------- kernel 2: fold edge_W into query_W --------------------
// Wfold[n][0:128]   = qW[n][0:128]                (raw part)
// Wfold[n][128:192] = sum_d qW[n][128+d]*eW[d][e] (edge part, composed)
// Wfold[n][192:256] = qW[n][256:320]              (tenc part)
// qbf[n] = qb[n] + sum_d qW[n][128+d]*eb[d]
__global__ void sam_fold_kernel(const float* __restrict__ qW, const float* __restrict__ qb,
                                const float* __restrict__ eW, const float* __restrict__ eb,
                                unsigned short* __restrict__ Wf, float* __restrict__ qbf) {
  int n = threadIdx.x;
  if (n >= D_) return;
  const float* qrow = qW + (size_t)n * 320;
  unsigned short* wrow = Wf + (size_t)n * KFOLD;
  for (int c = 0; c < 128; ++c) wrow[c] = f2bf(qrow[c]);
  for (int e = 0; e < 64; ++e) {
    float s = 0.f;
    for (int d = 0; d < 128; ++d) s += qrow[128 + d] * eW[d * 64 + e];
    wrow[128 + e] = f2bf(s);
  }
  for (int j = 0; j < 64; ++j) wrow[192 + j] = f2bf(qrow[256 + j]);
  float bs = qb[n];
  for (int d = 0; d < 128; ++d) bs += qrow[128 + d] * eb[d];
  qbf[n] = bs;
}

// ------------------- kernel 3/4: fused SAM cell + scatter ------------------
__global__ __launch_bounds__(WAVES_PER_BLOCK * 32)
void sam_pass_kernel(const int*   __restrict__ nodes,
                     const float* __restrict__ efeat_in,  // edge_features [B,64]
                     const float* __restrict__ ctime,     // [B]
                     const float* __restrict__ rawmem,    // [N,128]
                     const float* __restrict__ protos,    // [N,5,128]
                     const unsigned short* __restrict__ Wf,  // [128,256] bf16
                     const float* __restrict__ qbf,          // [128]
                     const float* __restrict__ gW,  const float* __restrict__ gb,
                     const float* __restrict__ temp,
                     const float* __restrict__ lng, const float* __restrict__ lnb,
                     const float* __restrict__ tw,  const float* __restrict__ tb,
                     float* __restrict__ out, int Bsz) {
  // [ Wfold LDS stage : 128 x 264 bf16 ][ per-wave A/Q tiles : 4 x 8448 B ]
  __shared__ __align__(16) char smem[WF_LDS_BYTES + WAVES_PER_BLOCK * WAVE_BYTES];
  unsigned short* WfL = (unsigned short*)smem;

  const int lane = threadIdx.x & 31;
  const int wid  = threadIdx.x >> 5;
  unsigned short* A = (unsigned short*)(smem + WF_LDS_BYTES + wid * WAVE_BYTES); // [16][264] bf16
  float*          Q = (float*)(smem + WF_LDS_BYTES + wid * WAVE_BYTES);          // [16][132] f32
  const int tileBase = blockIdx.x * (WAVES_PER_BLOCK * 16) + wid * 16;

  // ---- stage Wfold into LDS (row pad 256 -> 264 for bank skew) -----------
  // 128 rows x 32 chunks of 16B; 4096 chunks over 128 threads
  {
    const int tid = threadIdx.x;
    for (int c = tid; c < D_ * 32; c += WAVES_PER_BLOCK * 32) {
      const int row = c >> 5, col8 = (c & 31) * 8;
      *(v4u*)&WfL[row * ASTRIDE + col8] = *(const v4u*)&Wf[row * KFOLD + col8];
    }
  }

  // hoisted per-lane constants
  const int   d0   = lane * 4;
  const int   j0   = lane * 2;
  const float twj0 = tw[j0],     twj1 = tw[j0 + 1];
  const float tbj0 = tb[j0],     tbj1 = tb[j0 + 1];

  // ---------------- phase 1: build bf16 A tile [16 x 256] in LDS ----------
  for (int r = 0; r < 16; ++r) {
    int m = tileBase + r; if (m > Bsz - 1) m = Bsz - 1;
    int node = nodes[m];
    // raw part (cols 0..127): float4 gather -> 4 bf16
    v4f rw = *(const v4f*)&rawmem[(size_t)node * D_ + d0];
    unsigned int p0 = (unsigned int)f2bf(rw.x) | ((unsigned int)f2bf(rw.y) << 16);
    unsigned int p1 = (unsigned int)f2bf(rw.z) | ((unsigned int)f2bf(rw.w) << 16);
    *(unsigned int*)&A[r * ASTRIDE + d0]     = p0;
    *(unsigned int*)&A[r * ASTRIDE + d0 + 2] = p1;
    // edge part (cols 128..191)
    v2f ef = *(const v2f*)&efeat_in[(size_t)m * 64 + j0];
    *(unsigned int*)&A[r * ASTRIDE + 128 + j0] =
        (unsigned int)f2bf(ef.x) | ((unsigned int)f2bf(ef.y) << 16);
    // tenc part (cols 192..255): cos(t*w + b)
    float ct = ctime[m];
    float t0 = __cosf(ct * twj0 + tbj0);
    float t1 = __cosf(ct * twj1 + tbj1);
    *(unsigned int*)&A[r * ASTRIDE + 192 + j0] =
        (unsigned int)f2bf(t0) | ((unsigned int)f2bf(t1) << 16);
  }
  __syncthreads();

  // ---------------- phase 2: WMMA GEMM  q = A[16x256] * Wfold^T -----------
  v8f acc[8] = {};
  const int arow = lane & 15;
  const int aoff = (lane >= 16) ? 8 : 0;    // A: interleaved {0..7,16..23}/{8..15,24..31}
  const int boff = (lane >= 16) ? 8 : 0;    // B: interleaved {0..7,16..23}/{8..15,24..31}
  const int ncol = lane & 15;
#pragma unroll
  for (int kt = 0; kt < 8; ++kt) {
    const int kb = kt * 32;
    BF16Frag a;
    a.u[0] = *(const v4u*)&A[arow * ASTRIDE + kb + aoff];
    a.u[1] = *(const v4u*)&A[arow * ASTRIDE + kb + aoff + 16];
    // load ALL B fragments for this K-tile first (distinct regs -> one wait,
    // then 8 back-to-back WMMAs instead of load->wait->wmma chains)
    BF16Frag b[8];
#pragma unroll
    for (int nt = 0; nt < 8; ++nt) {
      const unsigned short* wr = &WfL[(nt * 16 + ncol) * ASTRIDE + kb + boff];
      b[nt].u[0] = *(const v4u*)wr;
      b[nt].u[1] = *(const v4u*)(wr + 16);
    }
#pragma unroll
    for (int nt = 0; nt < 8; ++nt) {
      acc[nt] = __builtin_amdgcn_wmma_f32_16x16x32_bf16(
          false, a.v, false, b[nt].v, (short)0, acc[nt], false, false);
    }
  }
  __syncthreads();  // A tile dead; reuse region as Q

  // spill q (+ folded bias) to LDS as [16][128] f32
  const int rbase = (lane >= 16) ? 8 : 0;   // C/D: VGPR v = row v (lo half) / v+8 (hi half)
#pragma unroll
  for (int nt = 0; nt < 8; ++nt) {
    const int n = nt * 16 + ncol;
    const float bias = qbf[n];
#pragma unroll
    for (int v = 0; v < 8; ++v) Q[(v + rbase) * QSTRIDE + n] = acc[nt][v] + bias;
  }
  __syncthreads();

  // ---------------- phase 3: per-row epilogue (whole wave per row) --------
  const v4f g4  = *(const v4f*)&lng[d0];
  const v4f b4  = *(const v4f*)&lnb[d0];
  const v4f gwr = *(const v4f*)&gW[d0];          // gate weights: raw part
  const v4f gwc = *(const v4f*)&gW[128 + d0];    // gate weights: cand part
  const float gwt0 = gW[256 + j0], gwt1 = gW[256 + j0 + 1];
  const float rtemp = 1.f / temp[0];
  const float gb0   = gb[0];

  for (int r = 0; r < 16; ++r) {
    const int m = tileBase + r;
    int mc = m; if (mc > Bsz - 1) mc = Bsz - 1;
    const int node = nodes[mc];

    v4f q = *(const v4f*)&Q[r * QSTRIDE + d0];
    // layer norm 1
    float mu  = wsum32(q.x + q.y + q.z + q.w) * (1.f / 128.f);
    float x0 = q.x - mu, x1 = q.y - mu, x2 = q.z - mu, x3 = q.w - mu;
    float var = wsum32(x0 * x0 + x1 * x1 + x2 * x2 + x3 * x3) * (1.f / 128.f);
    float inv = rsqrtf(var + LN_EPS);
    float qn0 = x0 * inv * g4.x + b4.x, qn1 = x1 * inv * g4.y + b4.y;
    float qn2 = x2 * inv * g4.z + b4.z, qn3 = x3 * inv * g4.w + b4.w;
    // l2-normalize q
    float n2 = wsum32(qn0 * qn0 + qn1 * qn1 + qn2 * qn2 + qn3 * qn3);
    float rn = 1.f / fmaxf(sqrtf(n2), EPS_);
    float qh0 = qn0 * rn, qh1 = qn1 * rn, qh2 = qn2 * rn, qh3 = qn3 * rn;

    // cosine sim against 5 prototypes
    const float* pb = &protos[(size_t)node * KP_ * D_];
    v4f p[KP_];
    float sim[KP_];
    float mx = -1e30f;
#pragma unroll
    for (int k = 0; k < KP_; ++k) {
      p[k] = *(const v4f*)&pb[k * D_ + d0];
      float pp = p[k].x * p[k].x + p[k].y * p[k].y + p[k].z * p[k].z + p[k].w * p[k].w;
      float qp = qh0 * p[k].x + qh1 * p[k].y + qh2 * p[k].z + qh3 * p[k].w;
      float pn = wsum32(pp);
      float qd = wsum32(qp);
      sim[k] = (qd / fmaxf(sqrtf(pn), EPS_)) * rtemp;
      mx = fmaxf(mx, sim[k]);
    }
    // softmax over K=5, cand = att . protos
    float es = 0.f, e[KP_];
#pragma unroll
    for (int k = 0; k < KP_; ++k) { e[k] = __expf(sim[k] - mx); es += e[k]; }
    const float res = 1.f / es;
    float c0 = 0.f, c1 = 0.f, c2 = 0.f, c3 = 0.f;
#pragma unroll
    for (int k = 0; k < KP_; ++k) {
      float a = e[k] * res;
      c0 += a * p[k].x; c1 += a * p[k].y; c2 += a * p[k].z; c3 += a * p[k].w;
    }

    // gate = sigmoid([raw|cand|tenc] . gW + gb)
    v4f rw = *(const v4f*)&rawmem[(size_t)node * D_ + d0];
    float ct = ctime[mc];
    float t0 = __cosf(ct * twj0 + tbj0);
    float t1 = __cosf(ct * twj1 + tbj1);
    float gp = rw.x * gwr.x + rw.y * gwr.y + rw.z * gwr.z + rw.w * gwr.w
             + c0 * gwc.x + c1 * gwc.y + c2 * gwc.z + c3 * gwc.w
             + t0 * gwt0 + t1 * gwt1;
    float z = wsum32(gp) + gb0;
    float gate = 1.f / (1.f + __expf(-z));

    // update + layer norm 2
    float u0 = (1.f - gate) * rw.x + gate * c0;
    float u1 = (1.f - gate) * rw.y + gate * c1;
    float u2 = (1.f - gate) * rw.z + gate * c2;
    float u3 = (1.f - gate) * rw.w + gate * c3;
    float mu2 = wsum32(u0 + u1 + u2 + u3) * (1.f / 128.f);
    float y0 = u0 - mu2, y1 = u1 - mu2, y2 = u2 - mu2, y3 = u3 - mu2;
    float var2 = wsum32(y0 * y0 + y1 * y1 + y2 * y2 + y3 * y3) * (1.f / 128.f);
    float inv2 = rsqrtf(var2 + LN_EPS);
    if (m < Bsz) {
      v4f o = { y0 * inv2 * g4.x + b4.x, y1 * inv2 * g4.y + b4.y,
                y2 * inv2 * g4.z + b4.z, y3 * inv2 * g4.w + b4.w };
      *(v4f*)&out[(size_t)node * D_ + d0] = o;   // scatter
    }
  }
}

// ---------------------------------------------------------------------------
extern "C" void kernel_launch(void* const* d_in, const int* in_sizes, int n_in,
                              void* d_out, int out_size, void* d_ws, size_t ws_size,
                              hipStream_t stream) {
  const int*   srcn = (const int*)d_in[0];
  const int*   tgtn = (const int*)d_in[1];
  const float* ef   = (const float*)d_in[2];
  const float* ct   = (const float*)d_in[3];
  const float* raw  = (const float*)d_in[4];
  const float* pr   = (const float*)d_in[5];
  const float* eW   = (const float*)d_in[6];
  const float* eb   = (const float*)d_in[7];
  const float* qW   = (const float*)d_in[8];
  const float* qb   = (const float*)d_in[9];
  const float* gW   = (const float*)d_in[10];
  const float* gb   = (const float*)d_in[11];
  const float* temp = (const float*)d_in[12];
  const float* lng  = (const float*)d_in[13];
  const float* lnb  = (const float*)d_in[14];
  const float* tw   = (const float*)d_in[15];
  const float* tb   = (const float*)d_in[16];
  (void)n_in; (void)out_size; (void)ws_size;

  const int Bsz = in_sizes[0];
  const int Nn  = in_sizes[4] / D_;
  float* out = (float*)d_out;

  unsigned short* Wf  = (unsigned short*)d_ws;                       // 128*256*2 = 64 KB
  float*          qbf = (float*)((char*)d_ws + D_ * KFOLD * sizeof(unsigned short));

  // 1) out = raw_memory
  const int n4 = Nn * D_ / 4;
  int copyBlocks = (n4 + 255) / 256; if (copyBlocks > 16384) copyBlocks = 16384;
  sam_copy_kernel<<<copyBlocks, 256, 0, stream>>>((const v4f*)raw, (v4f*)out, n4);

  // 2) fold edge projection into query weights (bf16)
  sam_fold_kernel<<<1, 128, 0, stream>>>(qW, qb, eW, eb, Wf, qbf);

  // 3) source pass then 4) target pass (ordering => target wins collisions)
  const int rowsPerBlock = WAVES_PER_BLOCK * 16;
  const int blocks = (Bsz + rowsPerBlock - 1) / rowsPerBlock;
  sam_pass_kernel<<<blocks, WAVES_PER_BLOCK * 32, 0, stream>>>(
      srcn, ef, ct, raw, pr, Wf, qbf, gW, gb, temp, lng, lnb, tw, tb, out, Bsz);
  sam_pass_kernel<<<blocks, WAVES_PER_BLOCK * 32, 0, stream>>>(
      tgtn, ef, ct, raw, pr, Wf, qbf, gW, gb, temp, lng, lnb, tw, tb, out, Bsz);
}
